// SATMessagePassing_81406810128510
// MI455X (gfx1250) — matrix operand order repaired and checked by reading, not compiled, not run on hardware
//
#include <hip/hip_runtime.h>
#include <hip/hip_bf16.h>
#include <stdint.h>

#define NVV   20000
#define NCC   80000
#define NEE   240000
#define FIN   96
#define FOUT  64
#define NH    4
#define HCH   256
#define G3    192

typedef _Float16 v16h __attribute__((ext_vector_type(16)));
typedef float    v8f  __attribute__((ext_vector_type(8)));

union H2 { uint32_t u; _Float16 h[2]; };

// ---------------------------------------------------------------------------
// WMMA GEMM:  C[M,N] = A[M,K] @ W[N,K]^T + bias[N]
// A row-major, stride lda (supports strided h_old views); W row-major (N x K).
// Block = 256 threads (8 waves), output tile 32(M) x 128(N). Each wave owns a
// 16-column slice with two accumulators (M rows 0-15 / 16-31), reusing its B
// fragment twice per K-step. K consumed 32/step via LDS staged as packed f16
// (float2/float4 global loads -> ds_store_b32/b64, branchless N-tail guard).
// Fragments packed per CDNA5 ISA 7.12.2 layouts.
// Requires: M % 32 == 0, K % 32 == 0 (true for all call sites).
// ---------------------------------------------------------------------------
__global__ __launch_bounds__(256) void gemm_wmma_f16(
    const float* __restrict__ A, int lda,
    const float* __restrict__ W,
    const float* __restrict__ bias,
    float* __restrict__ C, int ldc,
    int M, int N, int K)
{
  __shared__ uint32_t As32[32 * 16];    // 32 rows x 32 halves (packed pairs)
  __shared__ uint32_t Bs32[128 * 16];   // 128 n-rows x 32 k-halves (packed)
  const _Float16* As = (const _Float16*)As32;
  const _Float16* Bs = (const _Float16*)Bs32;

  const int t    = threadIdx.x;
  const int lane = t & 31;
  const int wave = t >> 5;
  const int half = lane >> 4;   // 0: lanes 0-15, 1: lanes 16-31
  const int l16  = lane & 15;
  const int m0   = blockIdx.x * 32;
  const int n0   = blockIdx.y * 128 + wave * 16;
  const bool active = (n0 < N);   // wave-uniform (N=192 tail)

  v8f acc0 = {};
  v8f acc1 = {};

  for (int k0 = 0; k0 < K; k0 += 32) {
    // Stage A tile 32x32: 512 f16-pairs, 2 per thread (float2 -> b32)
    #pragma unroll
    for (int i = 0; i < 2; ++i) {
      int p  = t + i * 256;
      int m  = p >> 4;          // 16 pairs per row
      int kp = p & 15;
      const float2 f = *(const float2*)&A[(size_t)(m0 + m) * lda + (k0 + 2 * kp)];
      H2 u; u.h[0] = (_Float16)f.x; u.h[1] = (_Float16)f.y;
      As32[m * 16 + kp] = u.u;
    }
    // Stage B tile 128x32: 1024 float4, 4 per thread (float4 -> b64)
    #pragma unroll
    for (int i = 0; i < 4; ++i) {
      int q  = t + i * 256;
      int nl = q >> 3;          // 8 float4 per row
      int kq = q & 7;
      int ng = blockIdx.y * 128 + nl;
      const bool ok = (ng < N);
      int ngs = ok ? ng : 0;    // always-in-bounds load, branchless zero
      float4 f = *(const float4*)&W[(size_t)ngs * K + (k0 + 4 * kq)];
      f.x = ok ? f.x : 0.0f; f.y = ok ? f.y : 0.0f;
      f.z = ok ? f.z : 0.0f; f.w = ok ? f.w : 0.0f;
      H2 u0; u0.h[0] = (_Float16)f.x; u0.h[1] = (_Float16)f.y;
      H2 u1; u1.h[0] = (_Float16)f.z; u1.h[1] = (_Float16)f.w;
      uint2 pk; pk.x = u0.u; pk.y = u1.u;
      *(uint2*)&Bs32[nl * 16 + 2 * kq] = pk;
    }
    __syncthreads();

    if (active) {
      v16h a0, a1, b;
      // B 32x16 fragment: VGPR j holds K = (j&4?16:0) + half*8 + (j&3)*2, +1
      #pragma unroll
      for (int j = 0; j < 8; ++j) {
        int kb = ((j & 4) << 2) + half * 8 + ((j & 3) << 1);
        b[2 * j]     = Bs[(wave * 16 + l16) * 32 + kb];
        b[2 * j + 1] = Bs[(wave * 16 + l16) * 32 + kb + 1];
      }
      // A 16x32 fragments: VGPR j holds K = (j&4?16:0) + half*8 + (j&3)*2, +1
      #pragma unroll
      for (int j = 0; j < 8; ++j) {
        int kb = ((j & 4) << 2) + half * 8 + ((j & 3) << 1);
        a0[2 * j]     = As[l16 * 32 + kb];
        a0[2 * j + 1] = As[l16 * 32 + kb + 1];
        a1[2 * j]     = As[(16 + l16) * 32 + kb];
        a1[2 * j + 1] = As[(16 + l16) * 32 + kb + 1];
      }
      acc0 = __builtin_amdgcn_wmma_f32_16x16x32_f16(
          false, a0, false, b, (short)0, acc0, false, false);
      acc1 = __builtin_amdgcn_wmma_f32_16x16x32_f16(
          false, a1, false, b, (short)0, acc1, false, false);
    }
    __syncthreads();
  }

  if (active) {
    int ng = n0 + l16;
    float bv = bias[ng];
    // C/D layout: VGPR i -> row (tile_m0 + half*8 + i), col = ng
    #pragma unroll
    for (int i = 0; i < 8; ++i) {
      int row = m0 + half * 8 + i;
      C[(size_t)row * ldc + ng] = acc0[i] + bv;
      C[(size_t)(row + 16) * ldc + ng] = acc1[i] + bv;
    }
  }
}

// ---------------------------------------------------------------------------
__global__ void fill_u32(uint32_t* p, uint32_t v, int n) {
  int i = blockIdx.x * blockDim.x + threadIdx.x;
  if (i < n) p[i] = v;
}

__device__ inline void atomicMaxFloat(float* addr, float val) {
  if (val >= 0.0f) atomicMax((int*)addr, __float_as_int(val));
  else             atomicMin((unsigned int*)addr, (unsigned int)__float_as_int(val));
}

// One wave per edge: g = leaky(xl[src]+xr[dst]); e[h] = dot(g[h,:], att[h,:])
__global__ __launch_bounds__(256) void edge_score(
    const float* __restrict__ xl, const float* __restrict__ xr,
    const int* __restrict__ src, const int* __restrict__ dst,
    const float* __restrict__ att,
    float* __restrict__ escore, float* __restrict__ segmax, int E)
{
  int e    = blockIdx.x * 8 + (threadIdx.x >> 5);
  int lane = threadIdx.x & 31;
  if (e >= E) return;
  int s = src[e], d = dst[e];
  const float* pl = xl + (size_t)s * HCH;
  const float* pr = xr + (size_t)d * HCH;
  float acc = 0.0f;
  int c0 = lane * 8;                      // 8 contiguous channels; head = lane>>3
  #pragma unroll
  for (int i = 0; i < 8; ++i) {
    int c = c0 + i;
    float g = pl[c] + pr[c];
    g = (g > 0.0f) ? g : 0.2f * g;
    acc += g * att[c];
  }
  acc += __shfl_xor(acc, 1);
  acc += __shfl_xor(acc, 2);
  acc += __shfl_xor(acc, 4);              // sum within the 8-lane head group
  if ((lane & 7) == 0) {
    int h = lane >> 3;
    escore[(size_t)e * NH + h] = acc;
    atomicMaxFloat(&segmax[(size_t)d * NH + h], acc);
  }
}

// One thread per (edge, head): p = exp(e - max[dst]); segment-sum p
__global__ void edge_softmax(
    const float* __restrict__ escore, const float* __restrict__ segmax,
    const int* __restrict__ dst,
    float* __restrict__ pbuf, float* __restrict__ segsum, int E)
{
  int idx = blockIdx.x * blockDim.x + threadIdx.x;
  if (idx >= E * NH) return;
  int e = idx >> 2, h = idx & 3;
  int d = dst[e];
  float p = __expf(escore[idx] - segmax[(size_t)d * NH + h]);
  pbuf[idx] = p;
  atomicAdd(&segsum[(size_t)d * NH + h], p);
}

// One wave per edge: out[dst] += xl[src] * alpha
__global__ __launch_bounds__(256) void edge_aggregate(
    const float* __restrict__ xl,
    const int* __restrict__ src, const int* __restrict__ dst,
    const float* __restrict__ pbuf, const float* __restrict__ segsum,
    float* __restrict__ out, int E)
{
  int e    = blockIdx.x * 8 + (threadIdx.x >> 5);
  int lane = threadIdx.x & 31;
  if (e >= E) return;
  int s = src[e], d = dst[e];
  int h = lane >> 3;
  float alpha = pbuf[(size_t)e * NH + h] / segsum[(size_t)d * NH + h];
  const float* pl = xl + (size_t)s * HCH;
  float* po = out + (size_t)d * HCH;
  int c0 = lane * 8;
  #pragma unroll
  for (int i = 0; i < 8; ++i) {
    int c = c0 + i;
    atomicAdd(&po[c], pl[c] * alpha);
  }
}

// One wave per row: x += bias; LayerNorm over 256 features; in-place
__global__ __launch_bounds__(256) void ln_rows(
    float* __restrict__ m, const float* __restrict__ bias,
    const float* __restrict__ gamma, const float* __restrict__ beta, int rows)
{
  int r    = blockIdx.x * 8 + (threadIdx.x >> 5);
  int lane = threadIdx.x & 31;
  if (r >= rows) return;
  float* p = m + (size_t)r * HCH;
  float v[8]; float s = 0.0f;
  #pragma unroll
  for (int i = 0; i < 8; ++i) {
    int c = i * 32 + lane;
    v[i] = p[c] + bias[c];
    s += v[i];
  }
  #pragma unroll
  for (int o = 1; o < 32; o <<= 1) s += __shfl_xor(s, o);
  float mu = s * (1.0f / HCH);
  float q = 0.0f;
  #pragma unroll
  for (int i = 0; i < 8; ++i) { float d = v[i] - mu; q += d * d; }
  #pragma unroll
  for (int o = 1; o < 32; o <<= 1) q += __shfl_xor(q, o);
  float rstd = rsqrtf(q * (1.0f / HCH) + 1e-5f);
  #pragma unroll
  for (int i = 0; i < 8; ++i) {
    int c = i * 32 + lane;
    p[c] = (v[i] - mu) * rstd * gamma[c] + beta[c];
  }
}

// GRU gates: gi/gh are [rows,192] (bias already fused by the GEMMs)
__global__ void gru_gates(
    const float* __restrict__ gi, const float* __restrict__ gh,
    const float* __restrict__ hold, int ldh,
    float* __restrict__ out, int rows)
{
  int idx = blockIdx.x * blockDim.x + threadIdx.x;
  if (idx >= rows * FOUT) return;
  int r = idx >> 6, c = idx & 63;
  const float* pi = gi + (size_t)r * G3;
  const float* ph = gh + (size_t)r * G3;
  float rr = 1.0f / (1.0f + __expf(-(pi[c]       + ph[c])));
  float zz = 1.0f / (1.0f + __expf(-(pi[64 + c]  + ph[64 + c])));
  float nn = tanhf(pi[128 + c] + rr * ph[128 + c]);
  float h  = hold[(size_t)r * ldh + c];
  out[(size_t)r * FOUT + c] = (1.0f - zz) * nn + zz * h;
}

// ---------------------------------------------------------------------------
extern "C" void kernel_launch(void* const* d_in, const int* in_sizes, int n_in,
                              void* d_out, int out_size, void* d_ws, size_t ws_size,
                              hipStream_t stream) {
  (void)in_sizes; (void)n_in; (void)out_size; (void)ws_size;

  const float* x        = (const float*)d_in[0];
  const int*   v2c_src  = (const int*)d_in[1];
  const int*   v2c_dst  = (const int*)d_in[2];
  const int*   c2v_src  = (const int*)d_in[3];
  const int*   c2v_dst  = (const int*)d_in[4];
  /* d_in[5] = num_vars scalar (== NVV) */
  const float* v2c_Wl   = (const float*)d_in[6];
  const float* v2c_bl   = (const float*)d_in[7];
  const float* v2c_Wr   = (const float*)d_in[8];
  const float* v2c_br   = (const float*)d_in[9];
  const float* v2c_att  = (const float*)d_in[10];
  const float* v2c_bias = (const float*)d_in[11];
  const float* c_gamma  = (const float*)d_in[12];
  const float* c_beta   = (const float*)d_in[13];
  const float* c_Wih    = (const float*)d_in[14];
  const float* c_Whh    = (const float*)d_in[15];
  const float* c_bih    = (const float*)d_in[16];
  const float* c_bhh    = (const float*)d_in[17];
  const float* c2v_Wl   = (const float*)d_in[18];
  const float* c2v_bl   = (const float*)d_in[19];
  const float* c2v_Wr   = (const float*)d_in[20];
  const float* c2v_br   = (const float*)d_in[21];
  const float* c2v_att  = (const float*)d_in[22];
  const float* c2v_bias = (const float*)d_in[23];
  const float* v_gamma  = (const float*)d_in[24];
  const float* v_beta   = (const float*)d_in[25];
  const float* v_Wih    = (const float*)d_in[26];
  const float* v_Whh    = (const float*)d_in[27];
  const float* v_bih    = (const float*)d_in[28];
  const float* v_bhh    = (const float*)d_in[29];

  float* out = (float*)d_out;                 // [h_v_new (NV x 64); h_c_new (NC x 64)]
  float* h_c_new = out + (size_t)NVV * FOUT;

  // Workspace layout (floats); bufA/bufB reused as gi/gh after aggregation
  float* bufA   = (float*)d_ws;                       // max NC x 256
  float* bufB   = bufA   + (size_t)NCC * HCH;         // max NC x 256
  float* bufM   = bufB   + (size_t)NCC * HCH;         // max NC x 256
  float* escore = bufM   + (size_t)NCC * HCH;         // E x 4
  float* pbuf   = escore + (size_t)NEE * NH;          // E x 4
  float* segmax = pbuf   + (size_t)NEE * NH;          // max NC x 4
  float* segsum = segmax + (size_t)NCC * NH;          // max NC x 4

  auto gemm = [&](const float* A, int lda, const float* W, const float* bias,
                  float* C, int M, int N, int K) {
    dim3 grid(M / 32, (N + 127) / 128);
    hipLaunchKernelGGL(gemm_wmma_f16, grid, dim3(256), 0, stream,
                       A, lda, W, bias, C, N, M, N, K);
  };
  auto fill = [&](float* p, uint32_t v, size_t n) {
    hipLaunchKernelGGL(fill_u32, dim3((unsigned)((n + 255) / 256)), dim3(256), 0, stream,
                       (uint32_t*)p, v, (int)n);
  };

  const uint32_t NEG_INF = 0xFF800000u;

  // ======================= Stage A: vars -> clauses =======================
  gemm(x, FIN, v2c_Wl, v2c_bl, bufA, NVV, HCH, FIN);                    // xl_v
  gemm(x + (size_t)NVV * FIN, FIN, v2c_Wr, v2c_br, bufB, NCC, HCH, FIN);// xr_c
  fill(segmax, NEG_INF, (size_t)NCC * NH);
  fill(segsum, 0u, (size_t)NCC * NH);
  fill(bufM, 0u, (size_t)NCC * HCH);
  hipLaunchKernelGGL(edge_score, dim3(NEE / 8), dim3(256), 0, stream,
                     bufA, bufB, v2c_src, v2c_dst, v2c_att, escore, segmax, NEE);
  hipLaunchKernelGGL(edge_softmax, dim3((NEE * NH) / 256), dim3(256), 0, stream,
                     escore, segmax, v2c_dst, pbuf, segsum, NEE);
  hipLaunchKernelGGL(edge_aggregate, dim3(NEE / 8), dim3(256), 0, stream,
                     bufA, v2c_src, v2c_dst, pbuf, segsum, bufM, NEE);
  hipLaunchKernelGGL(ln_rows, dim3(NCC / 8), dim3(256), 0, stream,
                     bufM, v2c_bias, c_gamma, c_beta, NCC);
  // GRU (clauses): gi = m_c @ Wih^T + bih ; gh = h_c_old @ Whh^T + bhh
  gemm(bufM, HCH, c_Wih, c_bih, bufA, NCC, G3, HCH);
  gemm(x + (size_t)NVV * FIN + (FIN - FOUT), FIN, c_Whh, c_bhh, bufB, NCC, G3, FOUT);
  hipLaunchKernelGGL(gru_gates, dim3((NCC * FOUT) / 256), dim3(256), 0, stream,
                     bufA, bufB, x + (size_t)NVV * FIN + (FIN - FOUT), FIN,
                     h_c_new, NCC);

  // ======================= Stage B: clauses -> vars =======================
  gemm(h_c_new, FOUT, c2v_Wl, c2v_bl, bufA, NCC, HCH, FOUT);            // xl_c
  gemm(x, FIN, c2v_Wr, c2v_br, bufB, NVV, HCH, FIN);                    // xr_v
  fill(segmax, NEG_INF, (size_t)NVV * NH);
  fill(segsum, 0u, (size_t)NVV * NH);
  fill(bufM, 0u, (size_t)NVV * HCH);
  hipLaunchKernelGGL(edge_score, dim3(NEE / 8), dim3(256), 0, stream,
                     bufA, bufB, c2v_src, c2v_dst, c2v_att, escore, segmax, NEE);
  hipLaunchKernelGGL(edge_softmax, dim3((NEE * NH) / 256), dim3(256), 0, stream,
                     escore, segmax, c2v_dst, pbuf, segsum, NEE);
  hipLaunchKernelGGL(edge_aggregate, dim3(NEE / 8), dim3(256), 0, stream,
                     bufA, c2v_src, c2v_dst, pbuf, segsum, bufM, NEE);
  hipLaunchKernelGGL(ln_rows, dim3(NVV / 8), dim3(256), 0, stream,
                     bufM, c2v_bias, v_gamma, v_beta, NVV);
  // GRU (vars)
  gemm(bufM, HCH, v_Wih, v_bih, bufA, NVV, G3, HCH);
  gemm(x + (FIN - FOUT), FIN, v_Whh, v_bhh, bufB, NVV, G3, FOUT);
  hipLaunchKernelGGL(gru_gates, dim3((NVV * FOUT) / 256), dim3(256), 0, stream,
                     bufA, bufB, x + (FIN - FOUT), FIN, out, NVV);
}
